// LSTM_78735340470746
// MI455X (gfx1250) — matrix operand order repaired and checked by reading, not compiled, not run on hardware
//
#include <hip/hip_runtime.h>
#include <cstdint>

// ---------------------------------------------------------------------------
// Bidirectional LSTM for MI455X (gfx1250).  B=32, T=512, D=H=512, 4H=2048.
// Phase 0: x -> bf16 blob; W_ih/W_hh -> WMMA-B bf16 fragment blobs; lengths.
// Phase 1: gates[t, mt, ntile, lane, r] (pre-swizzled C-fragment layout, bf16)
//          = x[b, rot(t), :] @ W_ih^T + b_ih, M flattened as m' = t*32 + b so
//          Phase-1 C-tiles coincide with the recurrent per-step tiles.
// Phase 2: one workgroup per direction (16 waves on one WGP):
//          h,c recurrence with v_wmma_f32_16x16x32_bf16; h in LDS; gates
//          streamed per step via TDM (tensor_load_to_lds), double-buffered.
// Workspace: ~159 MB.
// ---------------------------------------------------------------------------

typedef __attribute__((ext_vector_type(16))) __bf16    v16bf;
typedef __attribute__((ext_vector_type(8)))  float     v8f;
typedef __attribute__((ext_vector_type(4)))  float     v4f;
typedef __attribute__((ext_vector_type(4)))  unsigned  v4u;
typedef __attribute__((ext_vector_type(8)))  int       v8i;
typedef __attribute__((ext_vector_type(4)))  int       v4i;

union BFrag { v16bf v; v4u q[2]; __bf16 b[16]; };
union H8    { v4u q;   __bf16 b[8]; };
union P4    { unsigned long long u; __bf16 b[4]; };

#define LSTM_B   32
#define LSTM_T   512
#define LSTM_H   512
#define LSTM_4H  2048
#define KTILES   16      // 512 / 32
#define NTILES   128     // 2048 / 16
#define STEPELEMS 65536  // 2*128*32*8 bf16 per timestep (128 KB)

__device__ __forceinline__ float sigm(float x) { return 1.0f / (1.0f + __expf(-x)); }
__device__ __forceinline__ float fast_tanh(float x) {
  float xc = fminf(15.0f, fmaxf(-15.0f, x));
  float e  = __expf(2.0f * xc);
  return (e - 1.0f) / (e + 1.0f);
}

// ---------------------------------------------------------------------------
// Phase 0a: lengths[b] = sum_t mask[b,t]
// ---------------------------------------------------------------------------
__global__ void lengths_kernel(const int* __restrict__ mask, int* __restrict__ lengths) {
  int b = threadIdx.x;
  if (b < LSTM_B) {
    int s = 0;
    for (int t = 0; t < LSTM_T; ++t) s += mask[b * LSTM_T + t];
    lengths[b] = s;
  }
}

// ---------------------------------------------------------------------------
// Phase 0b: x (f32, [B*T, 512]) -> bf16 blob, 4 elements per thread.
// ---------------------------------------------------------------------------
__global__ void pack_x_kernel(const float* __restrict__ x, __bf16* __restrict__ xb) {
  int i = (blockIdx.x * blockDim.x + threadIdx.x) * 4;
  v4f v = *(const v4f*)(x + i);
  P4 p;
#pragma unroll
  for (int j = 0; j < 4; ++j) p.b[j] = (__bf16)v[j];
  *(unsigned long long*)(xb + i) = p.u;
}

// ---------------------------------------------------------------------------
// Phase 0c: pack W [2048 x 512] f32 row-major into bf16 WMMA-B fragments.
// B-tile (K=32 x N=16): lane l holds n = l&15, k = (l>>4)*8 + (e&7) + (e>>3)*16, e=0..15.
// One thread per (ntile, kstep, lane): 65536 threads; 32B out each.
// ---------------------------------------------------------------------------
__global__ void pack_w_kernel(const float* __restrict__ W, __bf16* __restrict__ blob) {
  int gid  = blockIdx.x * blockDim.x + threadIdx.x;   // [0, 65536)
  int lane = gid & 31;
  int ks   = (gid >> 5) & 15;
  int nt   = gid >> 9;                                // [0, 128)
  int n    = nt * 16 + (lane & 15);
  int kb   = ks * 32 + (lane >> 4) * 8;
  BFrag f;
#pragma unroll
  for (int e = 0; e < 16; ++e) f.b[e] = (__bf16)(W[n * LSTM_H + kb + (e & 7) + (e >> 3) * 16]);
  __bf16* dst = blob + (size_t)gid * 16;
  *(v4u*)(dst)     = f.q[0];
  *(v4u*)(dst + 8) = f.q[1];
}

// ---------------------------------------------------------------------------
// Phase 1: GEMM with M flattened as m' = t*32 + b.
// One wave: one t, 2 M-tiles (b 0..15 / 16..31) x 4 N-tiles, k-loop of 16.
// Output stored pre-swizzled: gates[t][mt][ntile][lane][r] bf16; each lane's
// 8 accumulator rows are contiguous -> one b128 store per tile.
// 16384 wave-jobs = 2048 blocks x 8 waves.
// ---------------------------------------------------------------------------
__global__ __launch_bounds__(256) void gemm_gates_kernel(
    const __bf16* __restrict__ xb, const __bf16* __restrict__ wpack,
    const float* __restrict__ bias, const int* __restrict__ lengths,
    __bf16* __restrict__ gates, int rotate) {
  int lane = threadIdx.x & 31;
  int wid  = blockIdx.x * 8 + (threadIdx.x >> 5);     // [0, 16384)
  int t    = wid >> 5;                                // [0, 512)
  int ng   = wid & 31;                                // [0, 32) -> 64 cols

  // A-row source indices (per-batch rotation for the backward direction).
  int rowA[2];
#pragma unroll
  for (int mt = 0; mt < 2; ++mt) {
    int b = mt * 16 + (lane & 15);
    int trow = rotate ? ((t + lengths[b]) & 511) : t;
    rowA[mt] = b * LSTM_T + trow;
  }
  int koff = (lane >> 4) * 8;

  v8f acc[2][4];
#pragma unroll
  for (int mt = 0; mt < 2; ++mt)
#pragma unroll
    for (int nt = 0; nt < 4; ++nt)
#pragma unroll
      for (int r = 0; r < 8; ++r) acc[mt][nt][r] = 0.0f;

  for (int ks = 0; ks < KTILES; ++ks) {
    int kb = ks * 32;
    BFrag a[2];
#pragma unroll
    for (int mt = 0; mt < 2; ++mt) {
      const __bf16* p = xb + (size_t)rowA[mt] * LSTM_H + kb + koff;
      a[mt].q[0] = *(const v4u*)(p);
      a[mt].q[1] = *(const v4u*)(p + 16);
    }
#pragma unroll
    for (int nt = 0; nt < 4; ++nt) {
      int ntile = ng * 4 + nt;
      const __bf16* wp = wpack + ((size_t)(ntile * 16 + ks) * 32 + lane) * 16;
      BFrag bf_;
      bf_.q[0] = *(const v4u*)(wp);
      bf_.q[1] = *(const v4u*)(wp + 8);
#pragma unroll
      for (int mt = 0; mt < 2; ++mt)
        acc[mt][nt] = __builtin_amdgcn_wmma_f32_16x16x32_bf16(
            false, a[mt].v, false, bf_.v, (short)0, acc[mt][nt], false, false);
    }
  }

  // bias + pre-swizzled store (one b128 per tile per lane)
#pragma unroll
  for (int nt = 0; nt < 4; ++nt) {
    int ntile = ng * 4 + nt;
    float bv = bias[ntile * 16 + (lane & 15)];
#pragma unroll
    for (int mt = 0; mt < 2; ++mt) {
      H8 o;
#pragma unroll
      for (int r = 0; r < 8; ++r) o.b[r] = (__bf16)(acc[mt][nt][r] + bv);
      __bf16* dst = gates + (size_t)t * STEPELEMS +
                    ((size_t)(mt * NTILES + ntile) * 32 + lane) * 8;
      *(v4u*)dst = o.q;
    }
  }
}

// ---------------------------------------------------------------------------
// TDM: copy one timestep's gate block (128 KB contiguous bf16) into LDS.
// tile_dim0 is 16-bit, so split 65536 elements into 2 rows of 32768.
// 6-arg builtin form (clang-23 / therock-10.0).
// ---------------------------------------------------------------------------
__device__ __forceinline__ void tdm_load_gates(const __bf16* gsrc, int t,
                                               unsigned lds_off) {
  unsigned long long ga =
      (unsigned long long)(uintptr_t)gsrc + (unsigned long long)t * STEPELEMS * 2ull;
  v4u g0;
  g0[0] = 1u;                                   // count=1 (valid user descriptor)
  g0[1] = lds_off;                              // lds_addr (bytes)
  g0[2] = (unsigned)(ga & 0xffffffffull);       // global_addr[31:0]
  g0[3] = (unsigned)((ga >> 32) & 0x1ffffffull) // global_addr[56:32]
          | (2u << 30);                         // type = 2 ("image")
  v8i g1;
  g1[0] = (1 << 16);                            // data_size=1 (2 bytes)
  g1[1] = (int)((32768u & 0xffffu) << 16);      // tensor_dim0 lo16 (=32768)
  g1[2] = (int)(2u << 16);                      // dim0 hi=0 | tensor_dim1 lo16 (=2)
  g1[3] = (int)(32768u << 16);                  // dim1 hi=0 | tile_dim0 (=32768)
  g1[4] = 2;                                    // tile_dim1 (=2) | tile_dim2=0
  g1[5] = 32768;                                // tensor_dim0_stride lo32
  g1[6] = 0;
  g1[7] = 0;
  v4i z4 = {0, 0, 0, 0};
  v8i z8 = {0, 0, 0, 0, 0, 0, 0, 0};
  __builtin_amdgcn_tensor_load_to_lds(g0, g1, z4, z4, z8, 0);
}

// ---------------------------------------------------------------------------
// Phase 2: per-direction recurrence. grid = 2 blocks x 512 threads (16 waves).
// Wave w owns hidden cols [32w, 32w+32): tiles nt = 32*g + 2*w + jt,
// g in {i,f,g,o}, jt in {0,1}; 16 acc tiles x 16 k-steps = 256 WMMA/wave/step.
// Gates arrive pre-swizzled: init = 16 x ds_read_b128 per lane.
// ---------------------------------------------------------------------------
__global__ __launch_bounds__(512, 1) void lstm_recurrent_kernel(
    const __bf16* __restrict__ gates_f, const __bf16* __restrict__ gates_b,
    const __bf16* __restrict__ whh_f,  const __bf16* __restrict__ whh_b,
    const int* __restrict__ lengths, float* __restrict__ out) {
  __shared__ __bf16 h_bf[LSTM_B * LSTM_H];       // 32 KB
  __shared__ __bf16 gbuf[2][STEPELEMS];          // 2 x 128 KB
  __shared__ int Ls[LSTM_B];

  const int dir = blockIdx.x;
  const __bf16* gates = dir ? gates_b : gates_f;
  const __bf16* whh   = dir ? whh_b   : whh_f;

  const int tid = threadIdx.x, lane = tid & 31, w = tid >> 5;

  for (int i = tid; i < LSTM_B * LSTM_H; i += 512) h_bf[i] = (__bf16)0.0f;
  if (tid < LSTM_B) Ls[tid] = lengths[tid];

  // Prime the pipeline: gates for step s=0 into buffer 0.
  if (w == 0) {
    int t0 = dir ? (LSTM_T - 1) : 0;
    tdm_load_gates(gates, t0, (unsigned)(uintptr_t)(void*)&gbuf[0][0]);
    __builtin_amdgcn_s_wait_tensorcnt(0);
  }
  __syncthreads();

  v8f cst[2][2];
#pragma unroll
  for (int mt = 0; mt < 2; ++mt)
#pragma unroll
    for (int jt = 0; jt < 2; ++jt)
#pragma unroll
      for (int r = 0; r < 8; ++r) cst[mt][jt][r] = 0.0f;

  const int mrowoff = 8 * (lane >> 4);
  const int ncl = lane & 15;

  for (int s = 0; s < LSTM_T; ++s) {
    const int t = dir ? (LSTM_T - 1 - s) : s;

    // Prefetch next step's gates into the other buffer (overlaps the GEMM).
    if (w == 0 && s + 1 < LSTM_T) {
      int tn = dir ? (t - 1) : (t + 1);
      tdm_load_gates(gates, tn, (unsigned)(uintptr_t)(void*)&gbuf[(s + 1) & 1][0]);
    }

    const __bf16* G = &gbuf[s & 1][0];

    // C init = input-projection gates, pre-swizzled: one b128 per tile.
    v8f acc[2][4][2];
#pragma unroll
    for (int mt = 0; mt < 2; ++mt)
#pragma unroll
      for (int g = 0; g < 4; ++g)
#pragma unroll
        for (int jt = 0; jt < 2; ++jt) {
          int nt = 32 * g + 2 * w + jt;
          H8 h8;
          h8.q = *(const v4u*)(G + ((mt * NTILES + nt) * 32 + lane) * 8);
#pragma unroll
          for (int r = 0; r < 8; ++r) acc[mt][g][jt][r] = (float)h8.b[r];
        }

    // v += h @ W_hh^T  (16 k-steps of 32)
    for (int ks = 0; ks < KTILES; ++ks) {
      int kb = ks * 32;
      BFrag a[2];
#pragma unroll
      for (int mt = 0; mt < 2; ++mt) {
        const __bf16* hp = &h_bf[(mt * 16 + ncl) * LSTM_H + kb + mrowoff];
        a[mt].q[0] = *(const v4u*)(hp);
        a[mt].q[1] = *(const v4u*)(hp + 16);
      }
#pragma unroll
      for (int g = 0; g < 4; ++g)
#pragma unroll
        for (int jt = 0; jt < 2; ++jt) {
          int nt = 32 * g + 2 * w + jt;
          const __bf16* wp = whh + ((size_t)(nt * 16 + ks) * 32 + lane) * 16;
          BFrag bf_;
          bf_.q[0] = *(const v4u*)(wp);
          bf_.q[1] = *(const v4u*)(wp + 8);
#pragma unroll
          for (int mt = 0; mt < 2; ++mt)
            acc[mt][g][jt] = __builtin_amdgcn_wmma_f32_16x16x32_bf16(
                false, a[mt].v, false, bf_.v, (short)0, acc[mt][g][jt], false, false);
        }
    }

    // LSTM cell update.
    float hout[2][2][8];
#pragma unroll
    for (int mt = 0; mt < 2; ++mt)
#pragma unroll
      for (int jt = 0; jt < 2; ++jt)
#pragma unroll
        for (int r = 0; r < 8; ++r) {
          float iv = sigm(acc[mt][0][jt][r]);
          float fv = sigm(acc[mt][1][jt][r]);
          float gv = fast_tanh(acc[mt][2][jt][r]);
          float ov = sigm(acc[mt][3][jt][r]);
          float c  = fv * cst[mt][jt][r] + iv * gv;
          cst[mt][jt][r] = c;
          hout[mt][jt][r] = fast_tanh(c) * ov;
        }

    __syncthreads();  // everyone finished reading h(s-1) and gates(s)

    // Write new h (bf16 -> LDS) and the fp32 output.
#pragma unroll
    for (int mt = 0; mt < 2; ++mt)
#pragma unroll
      for (int jt = 0; jt < 2; ++jt)
#pragma unroll
        for (int r = 0; r < 8; ++r) {
          int brow = mt * 16 + r + mrowoff;
          int col  = w * 32 + jt * 16 + ncl;
          float hv = hout[mt][jt][r];
          h_bf[brow * LSTM_H + col] = (__bf16)hv;
          int tout = dir ? ((t + Ls[brow]) & 511) : t;   // undo backward rotation
          out[((size_t)brow * LSTM_T + tout) * (2 * LSTM_H) + dir * LSTM_H + col] = hv;
        }

    if (w == 0) __builtin_amdgcn_s_wait_tensorcnt(0);  // gates(s+1) landed
    __syncthreads();  // h(s) visible; gates(s+1) visible
  }
}

// ---------------------------------------------------------------------------
// Host-side launcher. Workspace layout (~159 MB):
//   gates_f (64MB) | gates_b (64MB) | whh_f_p | whh_b_p | wih_f_p | wih_b_p
//   (2MB each) | x_bf (16MB) | lengths (128B)
// ---------------------------------------------------------------------------
extern "C" void kernel_launch(void* const* d_in, const int* in_sizes, int n_in,
                              void* d_out, int out_size, void* d_ws, size_t ws_size,
                              hipStream_t stream) {
  const float* x     = (const float*)d_in[0];
  const int*   mask  = (const int*)d_in[1];
  const float* Wih_f = (const float*)d_in[2];
  const float* bih_f = (const float*)d_in[3];
  const float* Whh_f = (const float*)d_in[4];
  const float* Wih_b = (const float*)d_in[5];
  const float* bih_b = (const float*)d_in[6];
  const float* Whh_b = (const float*)d_in[7];
  float* out = (float*)d_out;

  char* ws = (char*)d_ws;
  size_t off = 0;
  const size_t gates_bytes = (size_t)LSTM_T * STEPELEMS * 2;   // 64 MB
  const size_t pack_bytes  = (size_t)LSTM_4H * LSTM_H * 2;     // 2 MB
  const size_t xb_bytes    = (size_t)LSTM_B * LSTM_T * LSTM_H * 2;  // 16 MB
  __bf16* gates_f = (__bf16*)(ws + off); off += gates_bytes;
  __bf16* gates_b = (__bf16*)(ws + off); off += gates_bytes;
  __bf16* whh_f_p = (__bf16*)(ws + off); off += pack_bytes;
  __bf16* whh_b_p = (__bf16*)(ws + off); off += pack_bytes;
  __bf16* wih_f_p = (__bf16*)(ws + off); off += pack_bytes;
  __bf16* wih_b_p = (__bf16*)(ws + off); off += pack_bytes;
  __bf16* x_bf    = (__bf16*)(ws + off); off += xb_bytes;
  int* lengths    = (int*)(ws + off);    off += 128;
  (void)in_sizes; (void)n_in; (void)out_size; (void)ws_size;

  hipLaunchKernelGGL(lengths_kernel, dim3(1), dim3(32), 0, stream, mask, lengths);

  const int xelems = LSTM_B * LSTM_T * LSTM_H;   // 8388608
  hipLaunchKernelGGL(pack_x_kernel, dim3(xelems / 4 / 256), dim3(256), 0, stream, x, x_bf);

  const int pack_threads = NTILES * KTILES * 32;  // 65536
  hipLaunchKernelGGL(pack_w_kernel, dim3(pack_threads / 256), dim3(256), 0, stream, Whh_f, whh_f_p);
  hipLaunchKernelGGL(pack_w_kernel, dim3(pack_threads / 256), dim3(256), 0, stream, Whh_b, whh_b_p);
  hipLaunchKernelGGL(pack_w_kernel, dim3(pack_threads / 256), dim3(256), 0, stream, Wih_f, wih_f_p);
  hipLaunchKernelGGL(pack_w_kernel, dim3(pack_threads / 256), dim3(256), 0, stream, Wih_b, wih_b_p);

  // 16384 wave-jobs, 8 waves (256 threads) per block.
  hipLaunchKernelGGL(gemm_gates_kernel, dim3(2048), dim3(256), 0, stream,
                     x_bf, wih_f_p, bih_f, lengths, gates_f, 0);
  hipLaunchKernelGGL(gemm_gates_kernel, dim3(2048), dim3(256), 0, stream,
                     x_bf, wih_b_p, bih_b, lengths, gates_b, 1);

  hipLaunchKernelGGL(lstm_recurrent_kernel, dim3(2), dim3(512), 0, stream,
                     gates_f, gates_b, whh_f_p, whh_b_p, lengths, out);
}
